// Attention_layer_19980187861062
// MI455X (gfx1250) — matrix operand order repaired and checked by reading, not compiled, not run on hardware
//
#include <hip/hip_runtime.h>
#include <hip/hip_bf16.h>

// ---------------------------------------------------------------------------
// MI455X (gfx1250) flash-attention pipeline, all matmuls on
// v_wmma_f32_16x16x32_bf16 (wave32 WMMA, f32 accumulate).
//
// Roofline: ~22.5 GFLOP total; materializing [8,4096,4096] scores would move
// ~1.5GB/stage through memory -> flash attention instead. Q/K/V/x_cat are
// 2MB bf16 each (L2-resident on a 192MB-L2 part), so the attention phase is
// compute-bound on the WMMA pipe. Contiguous tile loads are vectorized to
// b128; f32->bf16 uses the native hardware convert (single VALU op) instead
// of a software RNE sequence.
// ---------------------------------------------------------------------------

typedef __attribute__((ext_vector_type(16))) __bf16 v16bf;
typedef __attribute__((ext_vector_type(8)))  __bf16 v8bf;
typedef __attribute__((ext_vector_type(8)))  float  v8f;
typedef __attribute__((ext_vector_type(4)))  float  v4f;

#define NROWS 4096
#define HDIM  256   // H*head_dim = 8*32

__device__ __forceinline__ __bf16 f2bf(float f) {
    // native fptrunc: gfx1250 has hardware BF16 converts
    return (__bf16)f;
}

// A-operand (16x32 bf16) per-lane fragment: 16 contiguous elements at
// rowptr[2*k8..2*k8+15]  (k8 = 8*(lane>>4)).
__device__ __forceinline__ v16bf load_a_bf16(const __bf16* rowptr, int k8) {
    v8bf lo = *(const v8bf*)(rowptr + 2 * k8);
    v8bf hi = *(const v8bf*)(rowptr + 2 * k8 + 8);
    return __builtin_shufflevector(lo, hi, 0, 1, 2, 3, 4, 5, 6, 7,
                                           8, 9, 10, 11, 12, 13, 14, 15);
}

// B-operand fragment from 32B-contiguous memory (e.g. K^T tiles): one
// 32-byte load per lane (2x global_load_b128).
__device__ __forceinline__ v16bf load_b_contig_bf16(const __bf16* p) {
    v8bf lo = *(const v8bf*)(p);
    v8bf hi = *(const v8bf*)(p + 8);
    return __builtin_shufflevector(lo, hi, 0, 1, 2, 3, 4, 5, 6, 7,
                                           8, 9, 10, 11, 12, 13, 14, 15);
}

// A-operand built from f32 memory (4x float4 loads + cvt).
__device__ __forceinline__ v16bf load_a_f32(const float* base, int k8) {
    v4f a0 = *(const v4f*)(base + 2 * k8);
    v4f a1 = *(const v4f*)(base + 2 * k8 + 4);
    v4f a2 = *(const v4f*)(base + 2 * k8 + 8);
    v4f a3 = *(const v4f*)(base + 2 * k8 + 12);
    v16bf a;
#pragma unroll
    for (int i = 0; i < 4; ++i) {
        a[i]      = f2bf(a0[i]);
        a[4 + i]  = f2bf(a1[i]);
        a[8 + i]  = f2bf(a2[i]);
        a[12 + i] = f2bf(a3[i]);
    }
    return a;
}

// ===========================================================================
// Kernel 1: fused concat + projection GEMM.
//   out_bf16[N][256] = concat(x0, x1)[N][Kdim] @ W (W: [8][Kdim][32],
//   output column col = h*32+k) + bias[256].
// One wave computes a 16-row x 32-col tile; K loop in steps of 32.
// ===========================================================================
__global__ void __launch_bounds__(128)
qkv_gemm_kernel(const float* __restrict__ x0,   // [N,256]
                const float* __restrict__ x1,   // [N,768] or unused
                const float* __restrict__ W,    // [8][Kdim][32]
                const float* __restrict__ bias, // [256] (H*32, head-major)
                __bf16* __restrict__ out,       // [N,256] bf16
                int Kdim)
{
    const int lane  = threadIdx.x & 31;
    const int wave  = threadIdx.x >> 5;
    const int tile  = blockIdx.x * 4 + wave;   // 2048 tiles total
    const int strip = tile & 7;                // 8 strips of 32 cols (== head)
    const int rowt  = tile >> 3;               // 256 row tiles
    const int n0    = rowt * 16;
    const int col0  = strip * 32;
    const int half  = lane >> 4;
    const int ln    = lane & 15;
    const int k8    = half * 8;                // A-layout K sub-base
    const int kh    = half * 16;               // B-layout K sub-base

    // B column pointers are loop-invariant except for the kb*32 term.
    const int  c0 = col0 + ln;
    const int  c1 = col0 + 16 + ln;
    const float* wp0 = W + ((long)(c0 >> 5) * Kdim) * 32 + (c0 & 31);
    const float* wp1 = W + ((long)(c1 >> 5) * Kdim) * 32 + (c1 & 31);

    v8f acc0 = {}; v8f acc1 = {};

    for (int kb = 0; kb < Kdim; kb += 32) {
        // ---- A tile (16x32 bf16): rows n0.., K = kb..kb+31, from concat x
        const float* abase;
        if (kb < 256) abase = x0 + (long)(n0 + ln) * 256 + kb;
        else          abase = x1 + (long)(n0 + ln) * 768 + (kb - 256);
        v16bf a = load_a_f32(abase, k8);

        // ---- B tiles (32x16 bf16 each): W element (d, col); d-stride = 32 f32
        v16bf b0, b1;
#pragma unroll
        for (int i = 0; i < 16; ++i) b0[i] = f2bf(wp0[(long)(kb + kh + i) * 32]);
#pragma unroll
        for (int i = 0; i < 16; ++i) b1[i] = f2bf(wp1[(long)(kb + kh + i) * 32]);

        acc0 = __builtin_amdgcn_wmma_f32_16x16x32_bf16(false, a, false, b0,
                                                       (short)0, acc0, false, false);
        acc1 = __builtin_amdgcn_wmma_f32_16x16x32_bf16(false, a, false, b1,
                                                       (short)0, acc1, false, false);
    }

    // ---- bias + bf16 store (C layout: vgpr r -> row r + 8*half, col = lane&15)
    const float bias0 = bias[c0];
    const float bias1 = bias[c1];
#pragma unroll
    for (int r = 0; r < 8; ++r) {
        int row = n0 + r + 8 * half;
        out[(long)row * HDIM + c0] = f2bf(acc0[r] + bias0);
        out[(long)row * HDIM + c1] = f2bf(acc1[r] + bias1);
    }
}

// ===========================================================================
// Kernel 2: flash attention. One wave owns (head h, 16 query rows).
// Streams keys in blocks of 32: QK^T (2 WMMAs), online softmax (half-wave
// shfl_xor reductions; C-layout rows align with 16-lane halves), P goes
// C-layout -> LDS -> A-layout (s_wait_dscnt 0 between), then P@V (2 WMMAs).
// ===========================================================================
__global__ void __launch_bounds__(128)
flash_attn_kernel(const __bf16* __restrict__ Qb,  // [N,256] head-major
                  const __bf16* __restrict__ Kb,
                  const __bf16* __restrict__ Vb,
                  __bf16* __restrict__ Xc)        // [N,256] head-major concat
{
    __shared__ __bf16 pbuf[4][16 * 32];            // 1KB per wave
    const int lane = threadIdx.x & 31;
    const int wave = threadIdx.x >> 5;
    const int tile = blockIdx.x * 4 + wave;        // 2048 = 8 heads * 256 blocks
    const int h    = tile >> 8;
    const int rowt = tile & 255;
    const int n0   = rowt * 16;
    const int cb   = h * 32;
    const int half = lane >> 4;
    const int ln   = lane & 15;
    const int k8   = half * 8;
    const int kh   = half * 16;
    const float scale = 0.0625f;                   // 1/sqrt(HID_DIM=256)

    // ---- Q tile as WMMA A operand (persistent across key loop)
    v16bf qa = load_a_bf16(Qb + (long)(n0 + ln) * HDIM + cb, k8);

    float mrow[8], lrow[8];
#pragma unroll
    for (int r = 0; r < 8; ++r) { mrow[r] = -1e30f; lrow[r] = 0.f; }
    v8f acc0 = {}, acc1 = {};
    __bf16* pt = pbuf[wave];

    for (int j = 0; j < NROWS; j += 32) {
        __builtin_prefetch(Kb + (long)(j + 32) * HDIM + cb, 0, 1);

        // ---- K^T as two B tiles (keys j..j+15, j+16..j+31); 32B/lane, b128
        v16bf kb0 = load_b_contig_bf16(Kb + (long)(j + ln) * HDIM + cb + kh);
        v16bf kb1 = load_b_contig_bf16(Kb + (long)(j + 16 + ln) * HDIM + cb + kh);

        v8f z = {};
        v8f s0 = __builtin_amdgcn_wmma_f32_16x16x32_bf16(false, qa, false, kb0,
                                                         (short)0, z, false, false);
        v8f s1 = __builtin_amdgcn_wmma_f32_16x16x32_bf16(false, qa, false, kb1,
                                                         (short)0, z, false, false);

        // ---- online softmax: row max over 32 keys (cross-lane in 16-lane half)
        float pm[8];
#pragma unroll
        for (int r = 0; r < 8; ++r) {
            s0[r] *= scale; s1[r] *= scale;
            pm[r] = fmaxf(s0[r], s1[r]);
        }
#pragma unroll
        for (int m = 1; m <= 8; m <<= 1) {
#pragma unroll
            for (int r = 0; r < 8; ++r) pm[r] = fmaxf(pm[r], __shfl_xor(pm[r], m));
        }

        float ps[8];
#pragma unroll
        for (int r = 0; r < 8; ++r) {
            float mn = fmaxf(mrow[r], pm[r]);
            float al = __expf(mrow[r] - mn);
            mrow[r] = mn;
            s0[r] = __expf(s0[r] - mn);
            s1[r] = __expf(s1[r] - mn);
            ps[r] = s0[r] + s1[r];
            lrow[r] *= al;
            acc0[r] *= al;
            acc1[r] *= al;
        }
#pragma unroll
        for (int m = 1; m <= 8; m <<= 1) {
#pragma unroll
            for (int r = 0; r < 8; ++r) ps[r] += __shfl_xor(ps[r], m);
        }
#pragma unroll
        for (int r = 0; r < 8; ++r) lrow[r] += ps[r];

        // ---- P: C layout -> LDS (wave-private) -> A layout
#pragma unroll
        for (int r = 0; r < 8; ++r) {
            pt[(r + 8 * half) * 32 + ln]      = f2bf(s0[r]);
            pt[(r + 8 * half) * 32 + 16 + ln] = f2bf(s1[r]);
        }
        asm volatile("s_wait_dscnt 0" ::: "memory");   // CDNA5 split DS counter
        v16bf pa = load_a_bf16(pt + ln * 32, k8);      // ds_load_b128 x2

        // ---- V as two B tiles (dims 0-15 and 16-31), K-dim = 32 keys.
        // stride-256 gather per lane, but coalesced across the 16-lane half.
        v16bf vb0, vb1;
#pragma unroll
        for (int i = 0; i < 16; ++i)
            vb0[i] = Vb[(long)(j + kh + i) * HDIM + cb + ln];
#pragma unroll
        for (int i = 0; i < 16; ++i)
            vb1[i] = Vb[(long)(j + kh + i) * HDIM + cb + 16 + ln];

        acc0 = __builtin_amdgcn_wmma_f32_16x16x32_bf16(false, pa, false, vb0,
                                                       (short)0, acc0, false, false);
        acc1 = __builtin_amdgcn_wmma_f32_16x16x32_bf16(false, pa, false, vb1,
                                                       (short)0, acc1, false, false);
    }

    // ---- normalize by row sum (NaN-guard: zero rows -> 0) and store bf16
#pragma unroll
    for (int r = 0; r < 8; ++r) {
        float inv = (lrow[r] > 0.f) ? (1.f / lrow[r]) : 0.f;
        int row = n0 + r + 8 * half;
        Xc[(long)row * HDIM + cb + ln]      = f2bf(acc0[r] * inv);
        Xc[(long)row * HDIM + cb + 16 + ln] = f2bf(acc1[r] * inv);
    }
}

// ===========================================================================
// Kernel 3: out[N,256] f32 = Xc(bf16) @ lin_w^T + lin_b.
//   B element (c, o) = lin_w[o*256 + c]  (contiguous along c -> b128 loads).
// ===========================================================================
__global__ void __launch_bounds__(128)
out_gemm_kernel(const __bf16* __restrict__ Xc,
                const float* __restrict__ lin_w,   // [256,256]
                const float* __restrict__ lin_b,   // [256]
                float* __restrict__ out)           // [N,256] f32
{
    const int lane  = threadIdx.x & 31;
    const int wave  = threadIdx.x >> 5;
    const int tile  = blockIdx.x * 4 + wave;
    const int strip = tile & 7;
    const int rowt  = tile >> 3;
    const int n0    = rowt * 16;
    const int col0  = strip * 32;
    const int half  = lane >> 4;
    const int ln    = lane & 15;
    const int k8    = half * 8;
    const int kh    = half * 16;

    v8f acc0 = {}, acc1 = {};
    for (int kb = 0; kb < 256; kb += 32) {
        v16bf a = load_a_bf16(Xc + (long)(n0 + ln) * HDIM + kb, k8);

        // B tiles: 16 contiguous f32 per lane -> 4x global_load_b128 each
        v16bf b0, b1;
        {
            const float* wp = lin_w + (long)(col0 + ln) * 256 + kb + kh;
            v4f w0 = *(const v4f*)(wp);
            v4f w1 = *(const v4f*)(wp + 4);
            v4f w2 = *(const v4f*)(wp + 8);
            v4f w3 = *(const v4f*)(wp + 12);
#pragma unroll
            for (int i = 0; i < 4; ++i) {
                b0[i] = f2bf(w0[i]); b0[4 + i] = f2bf(w1[i]);
                b0[8 + i] = f2bf(w2[i]); b0[12 + i] = f2bf(w3[i]);
            }
            wp = lin_w + (long)(col0 + 16 + ln) * 256 + kb + kh;
            w0 = *(const v4f*)(wp);
            w1 = *(const v4f*)(wp + 4);
            w2 = *(const v4f*)(wp + 8);
            w3 = *(const v4f*)(wp + 12);
#pragma unroll
            for (int i = 0; i < 4; ++i) {
                b1[i] = f2bf(w0[i]); b1[4 + i] = f2bf(w1[i]);
                b1[8 + i] = f2bf(w2[i]); b1[12 + i] = f2bf(w3[i]);
            }
        }

        acc0 = __builtin_amdgcn_wmma_f32_16x16x32_bf16(false, a, false, b0,
                                                       (short)0, acc0, false, false);
        acc1 = __builtin_amdgcn_wmma_f32_16x16x32_bf16(false, a, false, b1,
                                                       (short)0, acc1, false, false);
    }

    const float lb0 = lin_b[col0 + ln];
    const float lb1 = lin_b[col0 + 16 + ln];
#pragma unroll
    for (int r = 0; r < 8; ++r) {
        int row = n0 + r + 8 * half;
        out[(long)row * 256 + col0 + ln]      = acc0[r] + lb0;
        out[(long)row * 256 + col0 + 16 + ln] = acc1[r] + lb1;
    }
}

// ===========================================================================
// Host launcher
// ===========================================================================
extern "C" void kernel_launch(void* const* d_in, const int* in_sizes, int n_in,
                              void* d_out, int out_size, void* d_ws, size_t ws_size,
                              hipStream_t stream) {
    const float* input_x = (const float*)d_in[0];   // [4096,256]
    const float* pe_Q    = (const float*)d_in[1];   // [4096,768]
    const float* pe_K    = (const float*)d_in[2];   // [4096,768]
    /* d_in[3] = A (unused by reference) */
    const float* WQ      = (const float*)d_in[4];   // [8,1024,32]
    const float* WK      = (const float*)d_in[5];   // [8,1024,32]
    const float* WV      = (const float*)d_in[6];   // [8,256,32]
    const float* Q_bias  = (const float*)d_in[7];   // [8,1,32] -> flat [256]
    const float* K_bias  = (const float*)d_in[8];
    const float* V_bias  = (const float*)d_in[9];
    const float* lin_w   = (const float*)d_in[10];  // [256,256]
    const float* lin_b   = (const float*)d_in[11];  // [256]
    float* out = (float*)d_out;                     // [4096,256] f32

    // workspace: 4 bf16 buffers of [4096][256] = 2MB each (8MB total)
    char* ws = (char*)d_ws;
    __bf16* Qb = (__bf16*)(ws + 0);
    __bf16* Kb = (__bf16*)(ws + (size_t)2 * 1024 * 1024);
    __bf16* Vb = (__bf16*)(ws + (size_t)4 * 1024 * 1024);
    __bf16* Xc = (__bf16*)(ws + (size_t)6 * 1024 * 1024);

    dim3 blk(128);   // 4 waves/block (wave32)
    dim3 grd(512);   // 2048 wave-tiles

    qkv_gemm_kernel<<<grd, blk, 0, stream>>>(input_x, pe_Q, WQ, Q_bias, Qb, 1024);
    qkv_gemm_kernel<<<grd, blk, 0, stream>>>(input_x, pe_K, WK, K_bias, Kb, 1024);
    qkv_gemm_kernel<<<grd, blk, 0, stream>>>(input_x, input_x /*unused*/, WV, V_bias, Vb, 256);
    flash_attn_kernel<<<grd, blk, 0, stream>>>(Qb, Kb, Vb, Xc);
    out_gemm_kernel<<<grd, blk, 0, stream>>>(Xc, lin_w, lin_b, out);
}
